// SampleBlock_53085795779109
// MI455X (gfx1250) — matrix operand order repaired and checked by reading, not compiled, not run on hardware
//
#include <hip/hip_runtime.h>
#include <hip/hip_bf16.h>

typedef __attribute__((ext_vector_type(16))) __bf16 v16bf;
typedef __attribute__((ext_vector_type(8)))  __bf16 v8bf;
typedef __attribute__((ext_vector_type(2)))  __bf16 v2bf;
typedef __attribute__((ext_vector_type(8)))  float  v8f;

#define MT 32
#define NT 128
#define KT 32

__device__ __forceinline__ v16bf cat16(v8bf lo, v8bf hi) {
  return __builtin_shufflevector(lo, hi, 0, 1, 2, 3, 4, 5, 6, 7,
                                 8, 9, 10, 11, 12, 13, 14, 15);
}

// B-side element load at clamped coords (validity handled by caller's mask).
//  MODE 0: concat(cur_feat, sup_feat)   MODE 1: plain   MODE 2: relu-concat-upsample
template<int MODE, int CIN, int HIN, int WIN>
__device__ __forceinline__ float ldB(const float* __restrict__ p0,
                                     const float* __restrict__ p1,
                                     const float* __restrict__ p2,
                                     int b, int ci, int iyc, int ixc)
{
  if (MODE == 0) {
    const float* src = (ci < 64) ? p0 : p1;
    int c = ci & 63;
    return src[((size_t)(b * 64 + c) * HIN + iyc) * WIN + ixc];
  } else if (MODE == 1) {
    return p0[((size_t)(b * CIN + ci) * HIN + iyc) * WIN + ixc];
  } else {
    float v;
    if (ci < 256)
      v = p0[((size_t)(b * 256 + ci) * 128 + iyc) * 128 + ixc];
    else if (ci < 384)
      v = p1[((size_t)(b * 128 + (ci - 256)) * 64 + (iyc >> 1)) * 64 + (ixc >> 1)];
    else
      v = p2[((size_t)(b * 128 + (ci - 384)) * 32 + (iyc >> 2)) * 32 + (ixc >> 2)];
    return v > 0.0f ? v : 0.0f;
  }
}

// Implicit-GEMM 3x3 conv (pad=1) via V_WMMA_F32_16X16X32_BF16.
// K reordered as k' = tap*CIN + ci (order-invariant): no per-element div/mod,
// per-tile uniform padding mask, branchless staging.
// Global->register->LDS pipeline: tile t+1's global loads issue before tile t's
// WMMAs, hiding HBM/L2 latency behind the matrix pipe. Single LDS buffer.
// 256 threads = 8 wave32; tile MT(32) x NT(128); 2 wmma per wave per K-tile.
template<int MODE, bool RELU, bool BIAS, bool MGUARD,
         int CIN, int COUT, int HIN, int WIN, int HOUT, int WOUT, int STRIDE>
__global__ __launch_bounds__(256)
void conv3x3_wmma(const float* __restrict__ p0, const float* __restrict__ p1,
                  const float* __restrict__ p2, const float* __restrict__ wgt,
                  const float* __restrict__ bias, float* __restrict__ out)
{
  constexpr int K      = CIN * 9;
  constexpr int HW     = HOUT * WOUT;             // power of two
  constexpr int CPB    = CIN / KT;                // channel-blocks per tap (pow2)
  constexpr int NTILES = 9 * CPB;

  __shared__ alignas(16) __bf16 As[MT][KT + 8];   // [m][k], 80B rows (16B aligned)
  __shared__ alignas(16) __bf16 Bs[NT][KT + 8];   // [n][k] for b128 fragment loads

  const int tid  = threadIdx.x;
  const int n0   = blockIdx.x * NT;
  const int m0   = blockIdx.y * MT;
  const int lane = tid & 31;
  const int wid  = tid >> 5;
  const int msub = (wid & 1) * 16;
  const int nsub = (wid >> 1) * 32;
  const int l15  = lane & 15;
  const int hs   = lane >> 4;

  // B staging: nn fixed per thread; thread owns kk = kkh .. kkh+15
  const int nn  = tid & (NT - 1);
  const int kkh = (tid >> 7) * 16;
  const int n   = n0 + nn;
  const int bB  = n / HW;
  const int rem = n & (HW - 1);
  const int oy  = rem / WOUT;
  const int ox  = rem & (WOUT - 1);
  const int iy0 = oy * STRIDE - 1;
  const int ix0 = ox * STRIDE - 1;

  // A staging: thread owns rows {mmP, mmP+16} x cols {kkp, kkp+1}
  const int mmP = tid >> 4;          // 0..15
  const int kkp = (tid & 15) * 2;    // 0..30 even

  v8f acc[2] = {};
  float rB[16];
  float rA[4];
  float vmask = 0.0f;

  // ---- global gather of one K-tile into registers ----
  auto gload = [&](int t) {
    int tap = t / CPB;               // shift (CPB pow2)
    int cb  = (t % CPB) * KT;        // and+shl
    int kh  = tap / 3;               // scalar magic-div, once per tile
    int kw  = tap - kh * 3;
    int iy  = iy0 + kh;
    int ix  = ix0 + kw;
    vmask = (iy >= 0 && iy < HIN && ix >= 0 && ix < WIN) ? 1.0f : 0.0f;
    int iyc = iy < 0 ? 0 : (iy > HIN - 1 ? HIN - 1 : iy);
    int ixc = ix < 0 ? 0 : (ix > WIN - 1 ? WIN - 1 : ix);
#pragma unroll
    for (int j = 0; j < 16; ++j)
      rB[j] = ldB<MODE, CIN, HIN, WIN>(p0, p1, p2, bB, cb + kkh + j, iyc, ixc);
#pragma unroll
    for (int r = 0; r < 2; ++r) {
#pragma unroll
      for (int j = 0; j < 2; ++j) {
        int m = m0 + mmP + 16 * r;
        if (MGUARD) {
          int mc    = m < COUT - 1 ? m : COUT - 1;   // branchless guard
          float msk = m < COUT ? 1.0f : 0.0f;
          rA[2 * r + j] = wgt[(size_t)mc * K + (cb + kkp + j) * 9 + tap] * msk;
        } else {
          rA[2 * r + j] = wgt[(size_t)m * K + (cb + kkp + j) * 9 + tap];
        }
      }
    }
    // prefetch next channel-slice of this weight row (speculative, in-page)
    __builtin_prefetch(&wgt[(size_t)(m0 + mmP) * K + (cb + kkp) * 9 + tap + 9 * KT],
                       0, 1);
  };

  // ---- convert + store staged registers to LDS (packed b32 stores) ----
  auto lstore = [&]() {
#pragma unroll
    for (int j = 0; j < 16; j += 2) {
      v2bf pk;
      pk[0] = (__bf16)(rB[j] * vmask);
      pk[1] = (__bf16)(rB[j + 1] * vmask);
      *(v2bf*)&Bs[nn][kkh + j] = pk;
    }
#pragma unroll
    for (int r = 0; r < 2; ++r) {
      v2bf pk;
      pk[0] = (__bf16)rA[2 * r];
      pk[1] = (__bf16)rA[2 * r + 1];
      *(v2bf*)&As[mmP + 16 * r][kkp] = pk;
    }
  };

  gload(0);
#pragma unroll 1
  for (int t = 0; t < NTILES; ++t) {
    lstore();
    __syncthreads();                      // staging visible to all waves
    if (t + 1 < NTILES) gload(t + 1);     // long-latency loads fly over WMMAs

    // ---- fragments: 2x ds_load_b128 each (contiguous k = 16*hs + s, same for A and B) ----
    const __bf16* arow = &As[msub + l15][0];
    v16bf af = cat16(*(const v8bf*)(arow + hs * 16),
                     *(const v8bf*)(arow + hs * 16 + 8));
    const __bf16* brow0 = &Bs[nsub + l15][0];
    const __bf16* brow1 = &Bs[nsub + 16 + l15][0];
    v16bf bf0 = cat16(*(const v8bf*)(brow0 + hs * 16),
                      *(const v8bf*)(brow0 + hs * 16 + 8));
    v16bf bf1 = cat16(*(const v8bf*)(brow1 + hs * 16),
                      *(const v8bf*)(brow1 + hs * 16 + 8));

    acc[0] = __builtin_amdgcn_wmma_f32_16x16x32_bf16(
        false, af, false, bf0, (short)0, acc[0], false, false);
    acc[1] = __builtin_amdgcn_wmma_f32_16x16x32_bf16(
        false, af, false, bf1, (short)0, acc[1], false, false);

    __syncthreads();                      // frag reads done before next lstore
  }

  // ---- store: C/D layout — VGPR r holds M = hs*8 + r, N = l15 ----
#pragma unroll
  for (int t = 0; t < 2; ++t) {
    int ncol = n0 + nsub + t * 16 + l15;
    int b    = ncol / HW;
    int rm   = ncol & (HW - 1);
#pragma unroll
    for (int r = 0; r < 8; ++r) {
      int m = m0 + msub + (hs << 3) + r;
      if (!MGUARD || m < COUT) {
        float v = acc[t][r];
        if (BIAS) v += bias[m];
        if (RELU) v = v > 0.0f ? v : 0.0f;
        out[(size_t)(b * COUT + m) * HW + rm] = v;
      }
    }
  }
}

// Deformable conv sampling + per-group einsum. dg=4, Cg=16, C=64, H=W=128.
__global__ __launch_bounds__(256)
void deform_kernel(const float* __restrict__ sup, const float* __restrict__ ow,
                   const float* __restrict__ dcw, float* __restrict__ out, int Bn)
{
  const int H = 128, W = 128, HW = H * W;
  int gidx = blockIdx.x * blockDim.x + threadIdx.x;
  int total = Bn * HW * 4;
  if (gidx >= total) return;
  int ob  = gidx & 3;
  int pix = gidx >> 2;
  int b   = pix / HW;
  int rem = pix - b * HW;
  int y   = rem >> 7;
  int x   = rem & 127;
  int obase = ob * 16;

  float acc[16];
#pragma unroll
  for (int o = 0; o < 16; ++o) acc[o] = 0.0f;

  for (int g = 0; g < 4; ++g) {
    for (int k = 0; k < 9; ++k) {
      int offc = (g * 9 + k) * 2;
      float dy = ow[((b * 73) + offc) * HW + rem];
      float dx = ow[((b * 73) + offc + 1) * HW + rem];
      float py = (float)(y + (k / 3) - 1) + dy;
      float px = (float)(x + (k % 3) - 1) + dx;
      float fy = floorf(py), fx = floorf(px);
      int y0 = (int)fy, x0 = (int)fx;
      int y1 = y0 + 1, x1 = x0 + 1;
      float wy1 = py - fy, wx1 = px - fx;
      float wy0 = 1.0f - wy1, wx0 = 1.0f - wx1;
      bool vy0 = (y0 >= 0) && (y0 < H), vy1 = (y1 >= 0) && (y1 < H);
      bool vx0 = (x0 >= 0) && (x0 < W), vx1 = (x1 >= 0) && (x1 < W);
      float m00 = (vy0 && vx0) ? wy0 * wx0 : 0.0f;
      float m01 = (vy0 && vx1) ? wy0 * wx1 : 0.0f;
      float m10 = (vy1 && vx0) ? wy1 * wx0 : 0.0f;
      float m11 = (vy1 && vx1) ? wy1 * wx1 : 0.0f;
      int y0c = y0 < 0 ? 0 : (y0 > H - 1 ? H - 1 : y0);
      int y1c = y1 < 0 ? 0 : (y1 > H - 1 ? H - 1 : y1);
      int x0c = x0 < 0 ? 0 : (x0 > W - 1 ? W - 1 : x0);
      int x1c = x1 < 0 ? 0 : (x1 > W - 1 ? W - 1 : x1);
      for (int c = 0; c < 16; ++c) {
        const float* fm = sup + (size_t)((b * 64) + (g * 16 + c)) * HW;
        float v = m00 * fm[y0c * W + x0c] + m01 * fm[y0c * W + x1c] +
                  m10 * fm[y1c * W + x0c] + m11 * fm[y1c * W + x1c];
        const float* wp = dcw + (size_t)obase * 576 + g * 144 + c * 9 + k;
#pragma unroll
        for (int o = 0; o < 16; ++o) acc[o] += v * wp[o * 576];
      }
    }
  }
#pragma unroll
  for (int o = 0; o < 16; ++o)
    out[((size_t)(b * 64) + (obase + o)) * HW + rem] = acc[o];
}

__global__ void weight_relu_kernel(const float* __restrict__ ow,
                                   float* __restrict__ out, int Bn)
{
  const int HW = 16384;
  int i = blockIdx.x * blockDim.x + threadIdx.x;
  if (i >= Bn * HW) return;
  int b = i / HW, rem = i - b * HW;
  float v = ow[(b * 73 + 72) * HW + rem];
  out[i] = v > 0.0f ? v : 0.0f;
}

extern "C" void kernel_launch(void* const* d_in, const int* in_sizes, int n_in,
                              void* d_out, int out_size, void* d_ws, size_t ws_size,
                              hipStream_t stream) {
  (void)in_sizes; (void)n_in; (void)out_size; (void)ws_size;
  const float* cur    = (const float*)d_in[0];
  const float* sup    = (const float*)d_in[1];
  // d_in[2] = idx (int, ==0 in setup) -> stride-2 branch
  const float* off_w1 = (const float*)d_in[3];
  const float* off_w2 = (const float*)d_in[4];
  const float* off_w3 = (const float*)d_in[5];
  const float* lat_w  = (const float*)d_in[6];
  const float* lat_b  = (const float*)d_in[7];
  const float* pred_w = (const float*)d_in[8];
  const float* pred_b = (const float*)d_in[9];
  const float* dcn_w  = (const float*)d_in[10];

  const int B = 4;
  float* ws = (float*)d_ws;
  float* c1  = ws;                                    // 4*256*16384 f32
  float* c2  = c1  + (size_t)4 * 256 * 16384;         // 4*128*4096
  float* c3  = c2  + (size_t)4 * 128 * 4096;          // 4*128*1024
  float* lat = c3  + (size_t)4 * 128 * 1024;          // 4*256*16384
  float* owb = lat + (size_t)4 * 256 * 16384;         // 4*73*16384

  dim3 blk(256);
  // conv1: concat input, M=256, N=65536, K=1152
  conv3x3_wmma<0, false, false, false, 128, 256, 128, 128, 128, 128, 1>
      <<<dim3(512, 8), blk, 0, stream>>>(cur, sup, nullptr, off_w1, nullptr, c1);
  // conv2: stride2, M=128, N=16384, K=2304
  conv3x3_wmma<1, false, false, false, 256, 128, 128, 128, 64, 64, 2>
      <<<dim3(128, 4), blk, 0, stream>>>(c1, nullptr, nullptr, off_w2, nullptr, c2);
  // conv3: stride2, M=128, N=4096, K=1152
  conv3x3_wmma<1, false, false, false, 128, 128, 64, 64, 32, 32, 2>
      <<<dim3(32, 4), blk, 0, stream>>>(c2, nullptr, nullptr, off_w3, nullptr, c3);
  // lat: fused relu-concat-upsample input, bias+relu, M=256, N=65536, K=4608
  conv3x3_wmma<2, true, true, false, 512, 256, 128, 128, 128, 128, 1>
      <<<dim3(512, 8), blk, 0, stream>>>(c1, c2, c3, lat_w, lat_b, lat);
  // pred: M=73 (3 M-tiles, branchless clamp+mask guard), bias only
  conv3x3_wmma<1, false, true, true, 256, 73, 128, 128, 128, 128, 1>
      <<<dim3(512, 3), blk, 0, stream>>>(lat, nullptr, nullptr, pred_w, pred_b, owb);

  // deformable sampling + einsum -> d_out[0 : 4*64*128*128]
  int total = B * 16384 * 4;
  deform_kernel<<<(total + 255) / 256, 256, 0, stream>>>(
      sup, owb, dcn_w, (float*)d_out, B);
  // relu(weight) -> tail of d_out
  weight_relu_kernel<<<(B * 16384 + 255) / 256, 256, 0, stream>>>(
      owb, (float*)d_out + (size_t)B * 64 * 16384, B);
}
